// MultiHeadSelfAttention_59536836657997
// MI455X (gfx1250) — matrix operand order repaired and checked by reading, not compiled, not run on hardware
//
#include <hip/hip_runtime.h>
#include <hip/hip_bf16.h>

#define D_MODEL 512
#define NHEAD   8
#define D_HEAD  64
#define SEQ     2048
#define BATCH   4
#define NROWS   (BATCH * SEQ)      // 8192
#define QK_SCALE 0.125f            // 1/sqrt(64)
#define LOG2E    1.44269504088896f
#define WELEMS  (D_MODEL * D_MODEL) // 262144

#define FA_WAVES 4                 // waves per flash-attn block (share K/V via LDS)
#define KSTR 72                    // padded LDS row stride for 64-wide K rows (bf16)
#define VSTR 40                    // padded LDS row stride for 32-wide V rows (bf16)

typedef __attribute__((ext_vector_type(16))) __bf16 bf16x16;
typedef __attribute__((ext_vector_type(8)))  float  f32x8;

union FragAB { bf16x16 v; unsigned int u[8]; };

static __device__ __forceinline__ float fast_exp2(float x) {
  return __builtin_amdgcn_exp2f(x);   // v_exp_f32 (TRANS, co-executes with WMMA)
}

static __device__ __forceinline__ unsigned short f32_bf16(float f) {
  unsigned int u = __float_as_uint(f);
  u += 0x7FFFu + ((u >> 16) & 1u);      // round-to-nearest-even
  return (unsigned short)(u >> 16);
}
static __device__ __forceinline__ unsigned int pack_bf16(float lo, float hi) {
  return (unsigned int)f32_bf16(lo) | ((unsigned int)f32_bf16(hi) << 16);
}

// 16B global -> LDS async copy (ASYNCcnt-tracked; CDNA5 async path)
static __device__ __forceinline__ void async_b128_to_lds(unsigned int ldsOff,
                                                         const void* gptr) {
  asm volatile("global_load_async_to_lds_b128 %0, %1, off"
               :: "v"(ldsOff), "v"((unsigned long long)gptr) : "memory");
}
static __device__ __forceinline__ void wait_async0() {
  asm volatile("s_wait_asynccnt 0x0" ::: "memory");
}

// ---------------------------------------------------------------------------
// One-shot fp32 -> bf16 conversion (RNE).
// ---------------------------------------------------------------------------
__global__ __launch_bounds__(256)
void cvt_bf16_kernel(const float* __restrict__ in,
                     unsigned short* __restrict__ out, int n4) {
  const int idx = blockIdx.x * blockDim.x + threadIdx.x;
  if (idx < n4) {
    float4 f = ((const float4*)in)[idx];
    uint2 p;
    p.x = pack_bf16(f.x, f.y);
    p.y = pack_bf16(f.z, f.w);
    ((uint2*)out)[idx] = p;
  }
}

// ---------------------------------------------------------------------------
// QKV projection: out = x @ W^T + bias (x, W already bf16).
// transposeV==0 -> [b,h,l,dh]  (Q, K); transposeV==1 -> [b,h,dh,l]  (V)
// One wave computes a 32x64 tile: 2 A-frags x 4 B-frags = 8 WMMAs / K-step.
// ---------------------------------------------------------------------------
__global__ __launch_bounds__(256)
void qkv_proj_kernel(const unsigned short* __restrict__ xb,  // [8192,512] bf16
                     const unsigned short* __restrict__ Wb,  // [512,512]  bf16 (out,in)
                     const float* __restrict__ bias,         // [512] fp32
                     unsigned short* __restrict__ out,
                     int transposeV) {
  const int wave = threadIdx.x >> 5;
  const int lane = threadIdx.x & 31;
  const int hl   = lane >> 4;
  const int l16  = lane & 15;
  const int rowBase = (blockIdx.x * 8 + wave) * 32;
  const int colBase = blockIdx.y * 64;

  f32x8 acc[2][4] = {};
  for (int k0 = 0; k0 < D_MODEL; k0 += 32) {
    FragAB a[2];
#pragma unroll
    for (int mt = 0; mt < 2; ++mt) {
      const unsigned short* xr =
          xb + (size_t)(rowBase + mt * 16 + l16) * D_MODEL + k0 + hl * 16;
      if (k0 + 64 < D_MODEL) __builtin_prefetch(xr + 64, 0, 3);
#pragma unroll
      for (int v = 0; v < 8; ++v) {
        const int k = 2 * v;   // A layout K index
        a[mt].u[v] = *(const unsigned int*)(xr + k);
      }
    }
    FragAB wb[4];
#pragma unroll
    for (int nt = 0; nt < 4; ++nt) {
      const unsigned short* wr =
          Wb + (size_t)(colBase + nt * 16 + l16) * D_MODEL + k0 + hl * 16;
#pragma unroll
      for (int v = 0; v < 8; ++v)                         // B layout: K pair = 2v (+16*half)
        wb[nt].u[v] = *(const unsigned int*)(wr + 2 * v);
    }
#pragma unroll
    for (int mt = 0; mt < 2; ++mt)
#pragma unroll
      for (int nt = 0; nt < 4; ++nt)
        acc[mt][nt] = __builtin_amdgcn_wmma_f32_16x16x32_bf16(
            false, a[mt].v, false, wb[nt].v, (short)0, acc[mt][nt], false, false);
  }

#pragma unroll
  for (int nt = 0; nt < 4; ++nt) {
    const int e  = colBase + nt * 16 + l16;     // C layout: column = lane&15
    const int h  = e >> 6;
    const int dh = e & 63;
    const float bv = bias[e];
#pragma unroll
    for (int mt = 0; mt < 2; ++mt) {
#pragma unroll
      for (int r = 0; r < 8; ++r) {
        const int m  = rowBase + mt * 16 + hl * 8 + r;    // C layout: row = vgpr + 8*half
        const int bb = m >> 11, li = m & 2047;
        const float val = acc[mt][nt][r] + bv;
        size_t addr;
        if (transposeV)
          addr = ((size_t)((bb * NHEAD + h) * D_HEAD + dh)) * SEQ + li;
        else
          addr = ((size_t)((bb * NHEAD + h) * SEQ + li)) * D_HEAD + dh;
        out[addr] = f32_bf16(val);
      }
    }
  }
}

// ---------------------------------------------------------------------------
// Flash attention, cooperative: 4 waves per block, each owning a 32-row Q
// tile (128 q rows / block). K/V 32-wide chunks are staged in LDS with
// double-buffered global_load_async_to_lds_b128 (ASYNCcnt) and shared by all
// 4 waves -> 4x less L2 traffic. 16 WMMAs per wave per chunk. Online softmax
// on the C-fragment layout; P transposed through per-wave padded LDS tiles
// (same-wave LDS is in-order, so only a wave_barrier is needed).
// ---------------------------------------------------------------------------
__global__ __launch_bounds__(FA_WAVES * 32)
void flash_attn_kernel(const unsigned short* __restrict__ Q,   // [b,h,l,dh] bf16
                       const unsigned short* __restrict__ K,   // [b,h,l,dh] bf16
                       const unsigned short* __restrict__ Vt,  // [b,h,dh,l] bf16
                       unsigned short* __restrict__ ctx) {     // [b,l,512]  bf16
  __shared__ unsigned short kBuf[2][32 * KSTR];        // 2 x 4.5 KB
  __shared__ unsigned short vBuf[2][64 * VSTR];        // 2 x 5 KB
  __shared__ unsigned int  pTile[FA_WAVES][32 * 17];   // per-wave transpose tile

  const int tid  = threadIdx.x;
  const int wave = tid >> 5;
  const int lane = tid & 31;
  const int hl   = lane >> 4;
  const int l16  = lane & 15;
  const int h    = blockIdx.y;
  const int b    = blockIdx.z;

  const size_t bh = ((size_t)(b * NHEAD + h)) * SEQ * D_HEAD;
  const unsigned short* Qp = Q  + bh;
  const unsigned short* Kp = K  + bh;
  const unsigned short* Vp = Vt + bh;

  const int qRow0 = blockIdx.x * (FA_WAVES * 32) + wave * 32;

  // issue async loads for one 32-wide K/V chunk into buffer `buf`
  auto issue_chunk = [&](int buf, int kv0) {
#pragma unroll
    for (int i = 0; i < 2; ++i) {            // K: 32 rows x 64 dh = 256 x 16B segs
      const int seg = i * 128 + tid;
      const int row = seg >> 3, sc = seg & 7;
      async_b128_to_lds(
          (unsigned int)(size_t)&kBuf[buf][row * KSTR + sc * 8],
          Kp + (size_t)(kv0 + row) * D_HEAD + sc * 8);
    }
#pragma unroll
    for (int i = 0; i < 2; ++i) {            // V^T: 64 rows x 32 kv = 256 x 16B segs
      const int seg = i * 128 + tid;
      const int row = seg >> 2, sc = seg & 3;
      async_b128_to_lds(
          (unsigned int)(size_t)&vBuf[buf][row * VSTR + sc * 8],
          Vp + (size_t)row * SEQ + kv0 + sc * 8);
    }
  };

  // Q A-fragments: 2 row subtiles x 2 K=32 fragments covering d_head=64
  FragAB qa[2][2];
#pragma unroll
  for (int mt = 0; mt < 2; ++mt)
#pragma unroll
    for (int kk = 0; kk < 2; ++kk) {
      const unsigned short* qr =
          Qp + (size_t)(qRow0 + mt * 16 + l16) * D_HEAD + kk * 32 + hl * 16;
#pragma unroll
      for (int v = 0; v < 8; ++v) {
        const int k = 2 * v;
        qa[mt][kk].u[v] = *(const unsigned int*)(qr + k);
      }
    }

  float mrun[2][8], lrun[2][8];
  f32x8 o[2][4] = {};
#pragma unroll
  for (int mt = 0; mt < 2; ++mt)
#pragma unroll
    for (int r = 0; r < 8; ++r) { mrun[mt][r] = -1e30f; lrun[mt][r] = 0.0f; }

  issue_chunk(0, 0);
  wait_async0();
  __syncthreads();

  for (int kv0 = 0; kv0 < SEQ; kv0 += 32) {
    const int cur = (kv0 >> 5) & 1;
    if (kv0 + 32 < SEQ) issue_chunk(cur ^ 1, kv0 + 32);   // prefetch next chunk

    // ---- S = scale * Q @ K^T : 32x32 block as 2x2 16x16 C tiles ----
    float s[2][2][8];
#pragma unroll
    for (int t = 0; t < 2; ++t) {
      FragAB kb0, kb1;
      {
        const uint4* p0 = (const uint4*)&kBuf[cur][(t * 16 + l16) * KSTR + hl * 16];
        const uint4* p1 = (const uint4*)&kBuf[cur][(t * 16 + l16) * KSTR + 32 + hl * 16];
        uint4 a0 = p0[0], a1 = p0[1], b0 = p1[0], b1 = p1[1];
        kb0.u[0] = a0.x; kb0.u[1] = a0.y; kb0.u[2] = a0.z; kb0.u[3] = a0.w;
        kb0.u[4] = a1.x; kb0.u[5] = a1.y; kb0.u[6] = a1.z; kb0.u[7] = a1.w;
        kb1.u[0] = b0.x; kb1.u[1] = b0.y; kb1.u[2] = b0.z; kb1.u[3] = b0.w;
        kb1.u[4] = b1.x; kb1.u[5] = b1.y; kb1.u[6] = b1.z; kb1.u[7] = b1.w;
      }
#pragma unroll
      for (int mt = 0; mt < 2; ++mt) {
        f32x8 sc = {};
        sc = __builtin_amdgcn_wmma_f32_16x16x32_bf16(false, qa[mt][0].v, false, kb0.v, (short)0, sc, false, false);
        sc = __builtin_amdgcn_wmma_f32_16x16x32_bf16(false, qa[mt][1].v, false, kb1.v, (short)0, sc, false, false);
#pragma unroll
        for (int r = 0; r < 8; ++r) s[mt][t][r] = sc[r] * QK_SCALE;
      }
    }

    // ---- online softmax per row subtile (16-lane shfl_xor reductions) ----
#pragma unroll
    for (int mt = 0; mt < 2; ++mt) {
#pragma unroll
      for (int r = 0; r < 8; ++r) {
        float mx = fmaxf(s[mt][0][r], s[mt][1][r]);
#pragma unroll
        for (int off = 8; off >= 1; off >>= 1)
          mx = fmaxf(mx, __shfl_xor(mx, off, 32));
        const float mnew = fmaxf(mrun[mt][r], mx);
        const float corr = fast_exp2((mrun[mt][r] - mnew) * LOG2E);
        mrun[mt][r] = mnew;
        lrun[mt][r] *= corr;
#pragma unroll
        for (int j = 0; j < 4; ++j) o[mt][j][r] *= corr;
        const float p0 = fast_exp2((s[mt][0][r] - mnew) * LOG2E);
        const float p1 = fast_exp2((s[mt][1][r] - mnew) * LOG2E);
        s[mt][0][r] = p0; s[mt][1][r] = p1;
        float ps = p0 + p1;
#pragma unroll
        for (int off = 8; off >= 1; off >>= 1)
          ps += __shfl_xor(ps, off, 32);
        lrun[mt][r] += ps;
      }
    }

    // ---- transpose P (C layout -> A fragments) via per-wave LDS tile ----
    unsigned short* pt = (unsigned short*)pTile[wave];
#pragma unroll
    for (int mt = 0; mt < 2; ++mt)
#pragma unroll
      for (int t = 0; t < 2; ++t) {
        const int col = t * 16 + l16;
#pragma unroll
        for (int r = 0; r < 8; ++r)
          pt[(mt * 16 + hl * 8 + r) * 34 + col] = f32_bf16(s[mt][t][r]);
      }
    __builtin_amdgcn_wave_barrier();   // same-wave LDS is in-order; just stop reordering
    FragAB pa[2];
#pragma unroll
    for (int mt = 0; mt < 2; ++mt)
#pragma unroll
      for (int v = 0; v < 8; ++v) {
        const int k = 2 * v + hl * 16;
        pa[mt].u[v] = pTile[wave][(mt * 16 + l16) * 17 + (k >> 1)];
      }
    __builtin_amdgcn_wave_barrier();

    // ---- O += P @ V : V B-fragment from LDS, shared by both row subtiles ----
#pragma unroll
    for (int j = 0; j < 4; ++j) {
      FragAB vb;
      {
        const uint4* vp = (const uint4*)&vBuf[cur][(j * 16 + l16) * VSTR + hl * 16];
        uint4 w0 = vp[0], w1 = vp[1];
        vb.u[0] = w0.x; vb.u[1] = w0.y; vb.u[2] = w0.z; vb.u[3] = w0.w;
        vb.u[4] = w1.x; vb.u[5] = w1.y; vb.u[6] = w1.z; vb.u[7] = w1.w;
      }
#pragma unroll
      for (int mt = 0; mt < 2; ++mt)
        o[mt][j] = __builtin_amdgcn_wmma_f32_16x16x32_bf16(
            false, pa[mt].v, false, vb.v, (short)0, o[mt][j], false, false);
    }

    wait_async0();       // prefetched chunk resident before next iteration
    __syncthreads();     // all waves done reading current buffers
  }

  // ---- epilogue: normalize and scatter to ctx [b, l, h*64+dh] ----
#pragma unroll
  for (int mt = 0; mt < 2; ++mt)
#pragma unroll
    for (int r = 0; r < 8; ++r) {
      const int li = qRow0 + mt * 16 + hl * 8 + r;
      const float inv = 1.0f / lrun[mt][r];
#pragma unroll
      for (int j = 0; j < 4; ++j) {
        const int e = h * D_HEAD + j * 16 + l16;
        ctx[((size_t)(b * SEQ + li)) * D_MODEL + e] = f32_bf16(o[mt][j][r] * inv);
      }
    }
}

// ---------------------------------------------------------------------------
// Output projection: out(f32) = ctx(bf16) @ Wo^T + bo   (Wo pre-converted)
// ---------------------------------------------------------------------------
__global__ __launch_bounds__(256)
void out_proj_kernel(const unsigned short* __restrict__ ctx,
                     const unsigned short* __restrict__ Wob,  // bf16
                     const float* __restrict__ bo,
                     float* __restrict__ out) {
  const int wave = threadIdx.x >> 5;
  const int lane = threadIdx.x & 31;
  const int hl   = lane >> 4;
  const int l16  = lane & 15;
  const int rowBase = (blockIdx.x * 8 + wave) * 32;
  const int colBase = blockIdx.y * 64;

  f32x8 acc[2][4] = {};
  for (int k0 = 0; k0 < D_MODEL; k0 += 32) {
    FragAB a[2];
#pragma unroll
    for (int mt = 0; mt < 2; ++mt) {
      const unsigned short* xr =
          ctx + (size_t)(rowBase + mt * 16 + l16) * D_MODEL + k0 + hl * 16;
#pragma unroll
      for (int v = 0; v < 8; ++v) {
        const int k = 2 * v;
        a[mt].u[v] = *(const unsigned int*)(xr + k);
      }
    }
    FragAB wb[4];
#pragma unroll
    for (int nt = 0; nt < 4; ++nt) {
      const unsigned short* wr =
          Wob + (size_t)(colBase + nt * 16 + l16) * D_MODEL + k0 + hl * 16;
#pragma unroll
      for (int v = 0; v < 8; ++v)
        wb[nt].u[v] = *(const unsigned int*)(wr + 2 * v);
    }
#pragma unroll
    for (int mt = 0; mt < 2; ++mt)
#pragma unroll
      for (int nt = 0; nt < 4; ++nt)
        acc[mt][nt] = __builtin_amdgcn_wmma_f32_16x16x32_bf16(
            false, a[mt].v, false, wb[nt].v, (short)0, acc[mt][nt], false, false);
  }

#pragma unroll
  for (int nt = 0; nt < 4; ++nt) {
    const int e = colBase + nt * 16 + l16;
    const float bv = bo[e];
#pragma unroll
    for (int mt = 0; mt < 2; ++mt)
#pragma unroll
      for (int r = 0; r < 8; ++r) {
        const int m = rowBase + mt * 16 + hl * 8 + r;
        out[(size_t)m * D_MODEL + e] = acc[mt][nt][r] + bv;
      }
  }
}

// ---------------------------------------------------------------------------
extern "C" void kernel_launch(void* const* d_in, const int* in_sizes, int n_in,
                              void* d_out, int out_size, void* d_ws, size_t ws_size,
                              hipStream_t stream) {
  const float* x  = (const float*)d_in[0];
  const float* Wq = (const float*)d_in[1];
  const float* bq = (const float*)d_in[2];
  const float* Wk = (const float*)d_in[3];
  const float* bk = (const float*)d_in[4];
  const float* Wv = (const float*)d_in[5];
  const float* bv = (const float*)d_in[6];
  const float* Wo = (const float*)d_in[7];
  const float* bo = (const float*)d_in[8];

  // bf16 workspace layout (u16 units):
  //   Q | K | V^T | ctx  (each SEG = 4*8*2048*64 = 4Mi elems)
  //   x_bf16 (SEG) | Wq | Wk | Wv | Wo (WELEMS each)    total ~44 MB
  unsigned short* ws = (unsigned short*)d_ws;
  const size_t SEG = (size_t)BATCH * NHEAD * SEQ * D_HEAD;  // 4,194,304
  unsigned short* Qw  = ws;
  unsigned short* Kw  = ws + SEG;
  unsigned short* Vw  = ws + 2 * SEG;
  unsigned short* Cw  = ws + 3 * SEG;
  unsigned short* Xb  = ws + 4 * SEG;
  unsigned short* Wqb = ws + 5 * SEG;
  unsigned short* Wkb = Wqb + WELEMS;
  unsigned short* Wvb = Wkb + WELEMS;
  unsigned short* Wob = Wvb + WELEMS;

  // one-shot conversions (VALU cost paid once, not inside GEMM loops)
  cvt_bf16_kernel<<<(NROWS * D_MODEL / 4 + 255) / 256, 256, 0, stream>>>(x,  Xb,  NROWS * D_MODEL / 4);
  cvt_bf16_kernel<<<(WELEMS / 4 + 255) / 256, 256, 0, stream>>>(Wq, Wqb, WELEMS / 4);
  cvt_bf16_kernel<<<(WELEMS / 4 + 255) / 256, 256, 0, stream>>>(Wk, Wkb, WELEMS / 4);
  cvt_bf16_kernel<<<(WELEMS / 4 + 255) / 256, 256, 0, stream>>>(Wv, Wvb, WELEMS / 4);
  cvt_bf16_kernel<<<(WELEMS / 4 + 255) / 256, 256, 0, stream>>>(Wo, Wob, WELEMS / 4);

  dim3 gemmGrid(NROWS / 256, D_MODEL / 64);   // 32 x 8, 256 threads (8 waves)
  qkv_proj_kernel<<<gemmGrid, 256, 0, stream>>>(Xb, Wqb, bq, Qw, 0);
  qkv_proj_kernel<<<gemmGrid, 256, 0, stream>>>(Xb, Wkb, bk, Kw, 0);
  qkv_proj_kernel<<<gemmGrid, 256, 0, stream>>>(Xb, Wvb, bv, Vw, 1);

  flash_attn_kernel<<<dim3(SEQ / (FA_WAVES * 32), NHEAD, BATCH),
                      FA_WAVES * 32, 0, stream>>>(Qw, Kw, Vw, Cw);

  out_proj_kernel<<<gemmGrid, 256, 0, stream>>>(Cw, Wob, bo, (float*)d_out);
}
